// SelectiveSSM_59571196395603
// MI455X (gfx1250) — matrix operand order repaired and checked by reading, not compiled, not run on hardware
//
#include <hip/hip_runtime.h>
#include <hip/hip_bf16.h>

// ---------------- problem constants ----------------
#define D_MODEL 1024
#define D_STATE 16
#define D_CONV  4
#define D_INNER 2048            // D_MODEL * EXPAND
#define BATCH   2
#define SEQ     2048
#define MROWS   (BATCH*SEQ)     // 4096 tokens
#define N_IN    (2*D_INNER)     // 4096
#define NS      (2*D_STATE+1)   // 33

typedef __bf16 bf16;
typedef bf16  v8bf  __attribute__((ext_vector_type(8)));
typedef bf16  v16bf __attribute__((ext_vector_type(16)));
typedef float v8f   __attribute__((ext_vector_type(8)));

__device__ __forceinline__ float siluf(float v) { return v / (1.0f + __expf(-v)); }
__device__ __forceinline__ float softplusf(float v) { return v > 20.0f ? v : log1pf(__expf(v)); }

// ---------------- precision / layout prep ----------------
__global__ void k_cvt_f32_bf16(const float* __restrict__ src, bf16* __restrict__ dst, int n) {
    int i = blockIdx.x * blockDim.x + threadIdx.x;
    if (i < n) dst[i] = (bf16)src[i];
}

// W [K][N] f32 -> Wt [N][K] bf16   (so every WMMA B-fragment is a contiguous 32B lane load)
__global__ void k_transpose_bf16(const float* __restrict__ W, bf16* __restrict__ Wt, int K, int N) {
    int i = blockIdx.x * blockDim.x + threadIdx.x;
    if (i >= K * N) return;
    int n = i / K, k = i - n * K;
    Wt[(size_t)n * K + k] = (bf16)W[(size_t)k * N + n];
}

// ---------------- WMMA GEMM: C[M][N] = A[M][K](bf16,row) * Bt[N][K](bf16,K-major) ----------------
// One wave32 -> 16x64 output tile: 1 A-fragment reused over 4 B-fragments per K-step of 32.
__global__ void k_gemm_bf16(const bf16* __restrict__ A, const bf16* __restrict__ Bt,
                            float* __restrict__ C, int M, int N, int K) {
    const int lane   = threadIdx.x & 31;
    const int wid    = blockIdx.x * (blockDim.x >> 5) + (threadIdx.x >> 5);
    const int tilesN = N >> 6;
    if (wid >= (M >> 4) * tilesN) return;
    const int mBase = (wid / tilesN) << 4;
    const int n0    = (wid % tilesN) << 6;
    const int lm    = lane & 15;
    const int khalf = lane >> 4;

    // A-fragment lane base: row (mBase+lm), starting at K offset khalf*16 (contiguous, matches B)
    const bf16* aRow = A + (size_t)(mBase + lm) * K + khalf * 16;
    // B-fragment lane base: col (n0+sub*16+lm), starting at K offset khalf*16 (contiguous, matches A)
    const bf16* b0 = Bt + (size_t)(n0 +  0 + lm) * K + khalf * 16;
    const bf16* b1 = Bt + (size_t)(n0 + 16 + lm) * K + khalf * 16;
    const bf16* b2 = Bt + (size_t)(n0 + 32 + lm) * K + khalf * 16;
    const bf16* b3 = Bt + (size_t)(n0 + 48 + lm) * K + khalf * 16;

    v8f acc0 = {0.f,0.f,0.f,0.f,0.f,0.f,0.f,0.f};
    v8f acc1 = acc0, acc2 = acc0, acc3 = acc0;

    for (int k0 = 0; k0 < K; k0 += 32) {
        __builtin_prefetch(aRow + 512, 0, 0);   // -> global_prefetch_b8, ~16 K-steps ahead
        __builtin_prefetch(b0   + 512, 0, 0);

        v8bf alo = *(const v8bf*)(aRow);        // K = k0 + 16*khalf  .. +7   (16B)
        v8bf ahi = *(const v8bf*)(aRow + 8);    // K = k0+16*khalf+8 .. +7    (16B)
        v16bf afrag;
        #pragma unroll
        for (int i = 0; i < 8; ++i) { afrag[i] = alo[i]; afrag[8 + i] = ahi[i]; }

        v16bf f0 = *(const v16bf*)(b0);         // K = k0+16*khalf .. +15     (32B)
        v16bf f1 = *(const v16bf*)(b1);
        v16bf f2 = *(const v16bf*)(b2);
        v16bf f3 = *(const v16bf*)(b3);

        acc0 = __builtin_amdgcn_wmma_f32_16x16x32_bf16(false, afrag, false, f0, (short)0, acc0, false, false);
        acc1 = __builtin_amdgcn_wmma_f32_16x16x32_bf16(false, afrag, false, f1, (short)0, acc1, false, false);
        acc2 = __builtin_amdgcn_wmma_f32_16x16x32_bf16(false, afrag, false, f2, (short)0, acc2, false, false);
        acc3 = __builtin_amdgcn_wmma_f32_16x16x32_bf16(false, afrag, false, f3, (short)0, acc3, false, false);

        aRow += 32; b0 += 32; b1 += 32; b2 += 32; b3 += 32;
    }

    // C/D f32 layout: lane column n0+lm, VGPR r -> row mBase + 8*khalf + r
    float* cBase = C + (size_t)(mBase + khalf * 8) * N + n0 + lm;
    #pragma unroll
    for (int r = 0; r < 8; ++r) {
        cBase[(size_t)r * N +  0] = acc0[r];
        cBase[(size_t)r * N + 16] = acc1[r];
        cBase[(size_t)r * N + 32] = acc2[r];
        cBase[(size_t)r * N + 48] = acc3[r];
    }
}

// ---------------- depthwise causal conv (K=4) + SiLU ----------------
// xz: [MROWS][N_IN] f32 (cols 0..D_INNER-1 = x_branch) -> xc: [MROWS][D_INNER] f32
__global__ void k_conv_silu(const float* __restrict__ xz, const float* __restrict__ cw,
                            const float* __restrict__ cb, float* __restrict__ xc) {
    int tid = blockIdx.x * blockDim.x + threadIdx.x;
    if (tid >= MROWS * D_INNER) return;
    int row = tid / D_INNER, d = tid - row * D_INNER;
    int l = row & (SEQ - 1);
    float s = cb[d];
    #pragma unroll
    for (int j = 0; j < D_CONV; ++j) {
        int ls = l - (D_CONV - 1) + j;
        if (ls >= 0) s += xz[(size_t)(row - l + ls) * N_IN + d] * cw[d * D_CONV + j];
    }
    xc[(size_t)row * D_INNER + d] = siluf(s);
}

// ---------------- tiny projection: ssm[row][33] = xc[row][:] @ W_x[2048][33] ----------------
__global__ void k_ssm_proj(const float* __restrict__ xc, const float* __restrict__ Wx,
                           float* __restrict__ ssm) {
    __shared__ float red[256];
    const int row = blockIdx.x;
    const int t = threadIdx.x;
    float acc[NS];
    #pragma unroll
    for (int j = 0; j < NS; ++j) acc[j] = 0.f;
    for (int c = t; c < D_INNER; c += 256) {
        float v = xc[(size_t)row * D_INNER + c];
        const float* w = Wx + (size_t)c * NS;
        #pragma unroll
        for (int j = 0; j < NS; ++j) acc[j] += v * w[j];
    }
    for (int j = 0; j < NS; ++j) {
        red[t] = acc[j];
        __syncthreads();
        for (int s = 128; s > 0; s >>= 1) {
            if (t < s) red[t] += red[t + s];
            __syncthreads();
        }
        if (t == 0) ssm[(size_t)row * NS + j] = red[0];
        __syncthreads();
    }
}

// ---------------- selective scan + skip + gating, output bf16 for final WMMA GEMM ----------------
// one lane per (b, d, s); 16-lane shfl_xor reduce for sum_s h*C; lane s==0 writes gated y.
__global__ void k_scan(const float* __restrict__ ssm, const float* __restrict__ xc,
                       const float* __restrict__ xz, const float* __restrict__ A_log,
                       const float* __restrict__ dt_bias, const float* __restrict__ D_param,
                       bf16* __restrict__ yg) {
    int tid = blockIdx.x * blockDim.x + threadIdx.x;      // 65536 lanes
    int s  = tid & (D_STATE - 1);
    int bd = tid >> 4;
    int b  = bd >> 11;                                    // / D_INNER
    int d  = bd & (D_INNER - 1);
    float As  = -__expf(A_log[d * D_STATE + s]);
    float dtb = dt_bias[d];
    float Dp  = D_param[d];
    float h = 0.f;
    const int row0 = b * SEQ;
    for (int l = 0; l < SEQ; ++l) {
        int row = row0 + l;
        const float* sp = ssm + (size_t)row * NS;
        float Bv = sp[s], Cv = sp[D_STATE + s], dtr = sp[2 * D_STATE];
        float dt = softplusf(dtr + dtb);
        float xcv = xc[(size_t)row * D_INNER + d];
        h = h * __expf(dt * As) + (dt * xcv) * Bv;
        float p = h * Cv;
        p += __shfl_xor(p, 1, 32);
        p += __shfl_xor(p, 2, 32);
        p += __shfl_xor(p, 4, 32);
        p += __shfl_xor(p, 8, 32);
        if (s == 0) {
            float y  = p + xcv * Dp;
            float zv = xz[(size_t)row * N_IN + D_INNER + d];
            yg[(size_t)row * D_INNER + d] = (bf16)(y * siluf(zv));
        }
    }
}

// ---------------- workspace layout (all 256B aligned) ----------------
#define ALIGN256(x) (((x) + 255) & ~(size_t)255)
static const size_t SZ_XBF   = (size_t)MROWS * D_MODEL * 2;   //  8 MB  x bf16
static const size_t SZ_WINT  = (size_t)N_IN * D_MODEL * 2;    //  8 MB  W_in^T bf16
static const size_t SZ_WOUTT = (size_t)D_MODEL * D_INNER * 2; //  4 MB  W_out^T bf16
static const size_t SZ_XZ    = (size_t)MROWS * N_IN * 4;      // 64 MB  xz f32
static const size_t SZ_XC    = (size_t)MROWS * D_INNER * 4;   // 32 MB  xc f32
static const size_t SZ_SSM   = (size_t)MROWS * NS * 4;        // ~0.5MB
static const size_t SZ_YG    = (size_t)MROWS * D_INNER * 2;   // 16 MB  gated y bf16

extern "C" void kernel_launch(void* const* d_in, const int* in_sizes, int n_in,
                              void* d_out, int out_size, void* d_ws, size_t ws_size,
                              hipStream_t stream) {
    const float* x      = (const float*)d_in[0];
    const float* W_in   = (const float*)d_in[1];
    const float* conv_w = (const float*)d_in[2];
    const float* conv_b = (const float*)d_in[3];
    const float* W_x    = (const float*)d_in[4];
    const float* A_log  = (const float*)d_in[5];
    const float* D_par  = (const float*)d_in[6];
    const float* dt_b   = (const float*)d_in[7];
    const float* W_out  = (const float*)d_in[8];
    float* out = (float*)d_out;

    char* ws = (char*)d_ws;
    size_t o = 0;
    bf16*  x_bf   = (bf16*)(ws + o);  o = ALIGN256(o + SZ_XBF);
    bf16*  winT   = (bf16*)(ws + o);  o = ALIGN256(o + SZ_WINT);
    bf16*  woutT  = (bf16*)(ws + o);  o = ALIGN256(o + SZ_WOUTT);
    float* xz     = (float*)(ws + o); o = ALIGN256(o + SZ_XZ);
    float* xc     = (float*)(ws + o); o = ALIGN256(o + SZ_XC);
    float* ssm    = (float*)(ws + o); o = ALIGN256(o + SZ_SSM);
    bf16*  yg     = (bf16*)(ws + o);  o = ALIGN256(o + SZ_YG);
    (void)ws_size; (void)in_sizes; (void)n_in; (void)out_size;

    // 1) precision/layout prep
    {
        int n = MROWS * D_MODEL;
        k_cvt_f32_bf16<<<(n + 255) / 256, 256, 0, stream>>>(x, x_bf, n);
        int nw = D_MODEL * N_IN;
        k_transpose_bf16<<<(nw + 255) / 256, 256, 0, stream>>>(W_in, winT, D_MODEL, N_IN);
        int nw2 = D_INNER * D_MODEL;
        k_transpose_bf16<<<(nw2 + 255) / 256, 256, 0, stream>>>(W_out, woutT, D_INNER, D_MODEL);
    }
    // 2) in-projection GEMM: xz[4096][4096] = x_bf[4096][1024] @ W_in  (WMMA bf16)
    {
        int waves = (MROWS / 16) * (N_IN / 64);          // 16384
        k_gemm_bf16<<<waves / 8, 256, 0, stream>>>(x_bf, winT, xz, MROWS, N_IN, D_MODEL);
    }
    // 3) depthwise causal conv + SiLU
    {
        int n = MROWS * D_INNER;
        k_conv_silu<<<(n + 255) / 256, 256, 0, stream>>>(xz, conv_w, conv_b, xc);
    }
    // 4) x-projection (33-wide) via block reduction
    k_ssm_proj<<<MROWS, 256, 0, stream>>>(xc, W_x, ssm);
    // 5) selective scan + skip + gate -> bf16
    k_scan<<<(BATCH * D_INNER * D_STATE) / 256, 256, 0, stream>>>(ssm, xc, xz, A_log, dt_b, D_par, yg);
    // 6) out-projection GEMM: out[4096][1024] = yg[4096][2048] @ W_out  (WMMA bf16)
    {
        int waves = (MROWS / 16) * (D_MODEL / 64);       // 4096
        k_gemm_bf16<<<waves / 8, 256, 0, stream>>>(yg, woutT, out, MROWS, D_MODEL, D_INNER);
    }
}
